// CausalSelfAttention_20126216749545
// MI455X (gfx1250) — compile-verified, matrix operand order checked by reading, not run on hardware
//
#include <hip/hip_runtime.h>
#include <hip/hip_bf16.h>

// ---------------------------------------------------------------------------
// CausalSelfAttention (per-token 64x64 variant), MI455X / gfx1250
//   B=8, L=2048, D=1024, NH=16, HD=64
// Strategy: bf16 WMMA for the 4 big GEMMs (compute-bound, AI~260 F/B),
// fp32 VALU for the tiny per-token attention (3% of FLOPs, needs softmax
// precision; softmax axis is thread-local -> no cross-lane reductions).
// ---------------------------------------------------------------------------

typedef __attribute__((ext_vector_type(16))) __bf16 v16bf;
typedef __attribute__((ext_vector_type(8)))  __bf16 v8bf;
typedef __attribute__((ext_vector_type(4)))  __bf16 v4bf;
typedef __attribute__((ext_vector_type(8)))  float  v8f;

#define DMODEL 1024
#define NHEAD  16
#define HDIM   64
#define BLTOK  (8 * 2048)          // B*L = 16384 tokens

// ---------------- conversion kernels ----------------

__global__ __launch_bounds__(256) void k_f32_to_bf16(
    const float* __restrict__ in, __bf16* __restrict__ out, int n4) {
  int i = blockIdx.x * blockDim.x + threadIdx.x;
  if (i < n4) {
    float4 f = ((const float4*)in)[i];
    v4bf o;
    o[0] = (__bf16)f.x; o[1] = (__bf16)f.y;
    o[2] = (__bf16)f.z; o[3] = (__bf16)f.w;
    ((v4bf*)out)[i] = o;
  }
}

// W is (K x N) row-major fp32; produce N-major bf16 (out[n*K + k]) so the
// GEMM B-tile becomes a contiguous-K copy into LDS.
__global__ __launch_bounds__(256) void k_transpose_to_bf16(
    const float* __restrict__ in, __bf16* __restrict__ out) {
  int idx = blockIdx.x * blockDim.x + threadIdx.x;  // over D*D, coalesced read
  int k = idx >> 10;
  int n = idx & (DMODEL - 1);
  out[(size_t)n * DMODEL + k] = (__bf16)in[idx];
}

// ---------------- WMMA bf16 GEMM ----------------
//  C(M x 1024) = A(M x 1024) * W(1024 x 1024) + bias
//  A: bf16 row-major.  Bt: bf16, W transposed (N-major): Bt[n*1024 + k].
//  Block tile 128x128, K-stage 64. 8 waves; wave grid 2(M) x 4(N); each wave
//  owns 64x32 = 4x2 tiles of 16x16 -> 8 v8f accumulators.

#define BM 128
#define BN 128
#define BK 64
#define LDSA 72   // 64 + 8 bf16 pad (row stride 144 B, 16B aligned)
#define LDSB 72

template <bool BF16OUT>
__global__ __launch_bounds__(256) void k_gemm_bf16(
    const __bf16* __restrict__ A, const __bf16* __restrict__ Bt,
    const float* __restrict__ bias, void* __restrict__ Cout) {
  __shared__ __bf16 As[BM * LDSA];
  __shared__ __bf16 Bs[BN * LDSB];

  const int tid  = threadIdx.x;
  const int lane = tid & 31;
  const int w    = tid >> 5;
  const int wm   = (w & 1) * 64;   // wave M offset within block tile
  const int wn   = (w >> 1) * 32;  // wave N offset within block tile
  const int bm   = blockIdx.y * BM;
  const int bn   = blockIdx.x * BN;

  // tile-loader mapping: 8 threads cover one 64-elem K row (8 bf16 each)
  const int lr = tid >> 3;        // 0..31: row within a 32-row pass
  const int lc = (tid & 7) * 8;   // K element offset

  v8f acc[4][2];
#pragma unroll
  for (int mt = 0; mt < 4; ++mt)
#pragma unroll
    for (int nt = 0; nt < 2; ++nt)
#pragma unroll
      for (int e = 0; e < 8; ++e) acc[mt][nt][e] = 0.0f;

  const int r     = lane & 15;          // row/col within 16x16 fragment
  const int half8 = (lane >> 4) * 8;    // K sub-offset per ISA A/B layout

  for (int kb = 0; kb < DMODEL; kb += BK) {
    // stage A and B tiles (each thread: 4 x 16B per tile)
#pragma unroll
    for (int p = 0; p < 4; ++p) {
      int row = lr + p * 32;
      *(uint4*)&As[row * LDSA + lc] =
          *(const uint4*)&A[(size_t)(bm + row) * DMODEL + kb + lc];
      *(uint4*)&Bs[row * LDSB + lc] =
          *(const uint4*)&Bt[(size_t)(bn + row) * DMODEL + kb + lc];
    }
    if (kb + BK < DMODEL) {
      __builtin_prefetch(&A[(size_t)(bm + lr) * DMODEL + kb + BK + lc], 0, 1);
      __builtin_prefetch(&Bt[(size_t)(bn + lr) * DMODEL + kb + BK + lc], 0, 1);
    }
    __syncthreads();

#pragma unroll
    for (int kk = 0; kk < BK; kk += 32) {
      // A fragment (16x32 bf16): lane m=r, VGPR0-3 hold K=half8+0..7,
      // VGPR4-7 hold K=16+half8+0..7  -> two b128 LDS loads.
      v16bf af[4];
#pragma unroll
      for (int mt = 0; mt < 4; ++mt) {
        const __bf16* p0 = &As[(wm + mt * 16 + r) * LDSA + kk + half8];
        ((uint4*)&af[mt])[0] = *(const uint4*)p0;
        ((uint4*)&af[mt])[1] = *(const uint4*)(p0 + 16);
      }
      // B fragment (32x16): same layout with N in place of M (Bt is N-major).
      v16bf bfm[2];
#pragma unroll
      for (int nt = 0; nt < 2; ++nt) {
        const __bf16* p0 = &Bs[(wn + nt * 16 + r) * LDSB + kk + half8];
        ((uint4*)&bfm[nt])[0] = *(const uint4*)p0;
        ((uint4*)&bfm[nt])[1] = *(const uint4*)(p0 + 16);
      }
#pragma unroll
      for (int mt = 0; mt < 4; ++mt)
#pragma unroll
        for (int nt = 0; nt < 2; ++nt)
          acc[mt][nt] = __builtin_amdgcn_wmma_f32_16x16x32_bf16(
              false, af[mt], false, bfm[nt], (short)0, acc[mt][nt], false,
              false);
    }
    __syncthreads();
  }

  // epilogue: C/D layout -> lane n = lane&15, rows (lane>>4)*8 + e
  const int rhalf = (lane >> 4) * 8;
#pragma unroll
  for (int nt = 0; nt < 2; ++nt) {
    int col = bn + wn + nt * 16 + r;
    float bv = bias[col];
#pragma unroll
    for (int mt = 0; mt < 4; ++mt) {
      int row0 = bm + wm + mt * 16 + rhalf;
#pragma unroll
      for (int e = 0; e < 8; ++e) {
        float val = acc[mt][nt][e] + bv;
        if (BF16OUT)
          ((__bf16*)Cout)[(size_t)(row0 + e) * DMODEL + col] = (__bf16)val;
        else
          ((float*)Cout)[(size_t)(row0 + e) * DMODEL + col] = val;
      }
    }
  }
}

// ---------------- per-token attention ----------------
// For each token: q,k,v are (64 x 16); scores = q k^T / 4 + 1e-6 (64x64),
// mask = (j<=i && score!=0), row softmax, y = probs * v  (64 x 16).
// 4 tokens per 256-thread block; thread i owns score row i (softmax axis is
// the row -> fully thread-local). Scores kept in registers (unrolled).

__global__ __launch_bounds__(256) void k_attention(
    const __bf16* __restrict__ q, const __bf16* __restrict__ k,
    const __bf16* __restrict__ v, __bf16* __restrict__ y) {
  __shared__ float ks[4][HDIM][NHEAD];
  __shared__ float vs[4][HDIM][NHEAD];

  const int tid = threadIdx.x;
  const size_t tok0 = (size_t)blockIdx.x * 4;

  // stage k,v for 4 tokens: 4096 bf16 each = 512 chunks of 8
  float* ksf = &ks[0][0][0];
  float* vsf = &vs[0][0][0];
#pragma unroll
  for (int p = 0; p < 2; ++p) {
    int idx = tid + p * 256;
    v8bf tk = ((const v8bf*)(k + tok0 * DMODEL))[idx];
    v8bf tv = ((const v8bf*)(v + tok0 * DMODEL))[idx];
    float4 a, b;
    a.x = (float)tk[0]; a.y = (float)tk[1]; a.z = (float)tk[2]; a.w = (float)tk[3];
    b.x = (float)tk[4]; b.y = (float)tk[5]; b.z = (float)tk[6]; b.w = (float)tk[7];
    *(float4*)(ksf + idx * 8) = a;
    *(float4*)(ksf + idx * 8 + 4) = b;
    a.x = (float)tv[0]; a.y = (float)tv[1]; a.z = (float)tv[2]; a.w = (float)tv[3];
    b.x = (float)tv[4]; b.y = (float)tv[5]; b.z = (float)tv[6]; b.w = (float)tv[7];
    *(float4*)(vsf + idx * 8) = a;
    *(float4*)(vsf + idx * 8 + 4) = b;
  }
  __syncthreads();

  const int t = tid >> 6;   // local token
  const int i = tid & 63;   // score row

  float qi[NHEAD];
  {
    const v8bf* qp = (const v8bf*)(q + (tok0 + t) * DMODEL + (size_t)i * NHEAD);
    v8bf q0 = qp[0], q1 = qp[1];
#pragma unroll
    for (int h = 0; h < 8; ++h) {
      qi[h] = (float)q0[h];
      qi[8 + h] = (float)q1[h];
    }
  }

  float s[HDIM];
#pragma unroll
  for (int j = 0; j < HDIM; ++j) {
    float acc = 0.0f;
#pragma unroll
    for (int h = 0; h < NHEAD; ++h) acc = fmaf(qi[h], ks[t][j][h], acc);
    float sc = acc * 0.25f + 1e-6f;   // / sqrt(NH) + eps
    // mask = tril(values): keep iff lower-triangular AND value nonzero
    s[j] = ((j <= i) && (sc != 0.0f)) ? sc : -__builtin_inff();
  }

  float m = -__builtin_inff();
#pragma unroll
  for (int j = 0; j < HDIM; ++j) m = fmaxf(m, s[j]);
  float sum = 0.0f;
#pragma unroll
  for (int j = 0; j < HDIM; ++j) {
    float e = __expf(s[j] - m);
    s[j] = e;
    sum += e;
  }
  float rs = 1.0f / sum;

  float acc[NHEAD];
#pragma unroll
  for (int h = 0; h < NHEAD; ++h) acc[h] = 0.0f;
#pragma unroll
  for (int j = 0; j < HDIM; ++j) {
    float p = s[j];
#pragma unroll
    for (int h = 0; h < NHEAD; ++h) acc[h] = fmaf(p, vs[t][j][h], acc[h]);
  }

  __bf16* yo = y + (tok0 + t) * DMODEL + (size_t)i * NHEAD;
  v8bf o0, o1;
#pragma unroll
  for (int h = 0; h < 8; ++h) {
    o0[h] = (__bf16)(acc[h] * rs);
    o1[h] = (__bf16)(acc[8 + h] * rs);
  }
  ((v8bf*)yo)[0] = o0;
  ((v8bf*)yo)[1] = o1;
}

// ---------------- launch ----------------

extern "C" void kernel_launch(void* const* d_in, const int* in_sizes, int n_in,
                              void* d_out, int out_size, void* d_ws,
                              size_t ws_size, hipStream_t stream) {
  const float* x  = (const float*)d_in[0];
  const float* Wq = (const float*)d_in[1];
  const float* bq = (const float*)d_in[2];
  const float* Wk = (const float*)d_in[3];
  const float* bk = (const float*)d_in[4];
  const float* Wv = (const float*)d_in[5];
  const float* bv = (const float*)d_in[6];
  const float* Wo = (const float*)d_in[7];
  const float* bo = (const float*)d_in[8];
  float* out = (float*)d_out;

  char* ws = (char*)d_ws;
  size_t off = 0;
  auto alloc = [&](size_t bytes) -> void* {
    void* p = ws + off;
    off += (bytes + 255) & ~(size_t)255;
    return p;
  };
  __bf16* xb  = (__bf16*)alloc((size_t)BLTOK * DMODEL * 2);   // 32 MB
  __bf16* wtq = (__bf16*)alloc((size_t)DMODEL * DMODEL * 2);  // 2 MB
  __bf16* wtk = (__bf16*)alloc((size_t)DMODEL * DMODEL * 2);
  __bf16* wtv = (__bf16*)alloc((size_t)DMODEL * DMODEL * 2);
  __bf16* wto = (__bf16*)alloc((size_t)DMODEL * DMODEL * 2);
  __bf16* qb  = (__bf16*)alloc((size_t)BLTOK * DMODEL * 2);   // 32 MB
  __bf16* kb  = (__bf16*)alloc((size_t)BLTOK * DMODEL * 2);
  __bf16* vb  = (__bf16*)alloc((size_t)BLTOK * DMODEL * 2);
  __bf16* yb  = (__bf16*)alloc((size_t)BLTOK * DMODEL * 2);   // total ~168 MB

  // 1) precision conversion (bandwidth-trivial: ~120 MB at 23.3 TB/s)
  {
    int n4 = BLTOK * DMODEL / 4;
    k_f32_to_bf16<<<(n4 + 255) / 256, 256, 0, stream>>>(x, xb, n4);
    int wb = DMODEL * DMODEL / 256;
    k_transpose_to_bf16<<<wb, 256, 0, stream>>>(Wq, wtq);
    k_transpose_to_bf16<<<wb, 256, 0, stream>>>(Wk, wtk);
    k_transpose_to_bf16<<<wb, 256, 0, stream>>>(Wv, wtv);
    k_transpose_to_bf16<<<wb, 256, 0, stream>>>(Wo, wto);
  }

  // 2) QKV projections: (16384x1024)x(1024x1024) each, WMMA bf16
  dim3 gg(DMODEL / BN, BLTOK / BM);  // (8, 128)
  k_gemm_bf16<true><<<gg, 256, 0, stream>>>(xb, wtq, bq, qb);
  k_gemm_bf16<true><<<gg, 256, 0, stream>>>(xb, wtk, bk, kb);
  k_gemm_bf16<true><<<gg, 256, 0, stream>>>(xb, wtv, bv, vb);

  // 3) per-token 64x64 masked softmax attention
  k_attention<<<BLTOK / 4, 256, 0, stream>>>(qb, kb, vb, yb);

  // 4) output projection -> fp32 d_out (+ bias)
  k_gemm_bf16<false><<<gg, 256, 0, stream>>>(yb, wto, bo, out);
}
